// PhyloLM_66048007078167
// MI455X (gfx1250) — matrix operand order repaired and checked by reading, not compiled, not run on hardware
//
#include <hip/hip_runtime.h>
#include <hip/hip_bf16.h>
#include <cstdint>

// ---------------- problem constants (from reference) ----------------
#define Bc    2
#define Rr    20
#define Cc    128
#define Hh    256
#define NHh   8
#define Dd    32
#define Pp    190
#define H4    1024
#define NROWS (Bc*Pp*Cc)   // 48640 rows of the residual stream

typedef __attribute__((ext_vector_type(16))) _Float16 v16h;
typedef __attribute__((ext_vector_type(8)))  float    v8f;

union AFrag { v16h h; uint4 q[2]; };

__device__ __forceinline__ float wave_sum(float v) {
#pragma unroll
  for (int o = 16; o > 0; o >>= 1) v += __shfl_xor(v, o, 32);
  return v;
}
__device__ __forceinline__ float wave_max(float v) {
#pragma unroll
  for (int o = 16; o > 0; o >>= 1) v = fmaxf(v, __shfl_xor(v, o, 32));
  return v;
}
__device__ __forceinline__ float gelu_tanh(float x) {
  const float c0 = 0.7978845608028654f, c1 = 0.044715f;
  float x3 = x * x * x;
  return 0.5f * x * (1.0f + tanhf(c0 * (x + c1 * x3)));
}

// CDNA5 async global->LDS copy (ASYNCcnt-tracked DMA, no VGPR round trip).
// vdst operand = LDS byte offset in a VGPR; vaddr = 64-bit global address.
__device__ __forceinline__ void async_copy_b128(void* lds, const void* gptr) {
  uint32_t lo = (uint32_t)(uintptr_t)lds;   // low 32 bits of generic = LDS offset
  asm volatile("global_load_async_to_lds_b128 %0, %1, off"
               :: "v"(lo), "v"(gptr) : "memory");
}
__device__ __forceinline__ void wait_async0() {
  asm volatile("s_wait_asynccnt 0x0" ::: "memory");
}

// ---------------- embedding + pair row-sum: x[b,p,c,:] = emb[tok[b,i,c]] + emb[tok[b,j,c]] ----
__global__ __launch_bounds__(256) void embed_kernel(const int* __restrict__ tokens,
                                                    const float* __restrict__ emb,
                                                    float* __restrict__ xf) {
  int g = blockIdx.x * 256 + threadIdx.x;          // one thread per 4 floats
  if (g >= NROWS * (Hh / 4)) return;
  int row = g >> 6;
  int h = (g & 63) << 2;
  int c = row % Cc;
  int bp = row / Cc;
  int p = bp % Pp;
  int b = bp / Pp;
  // decode pair index p -> (i, j) with np.triu_indices ordering
  int i = 0, rem = p, span = Rr - 1;
  while (rem >= span) { rem -= span; ++i; --span; }
  int j = i + 1 + rem;
  int ti = tokens[(b * Rr + i) * Cc + c];
  int tj = tokens[(b * Rr + j) * Cc + c];
  float4 a = *(const float4*)&emb[(size_t)ti * Hh + h];
  float4 d = *(const float4*)&emb[(size_t)tj * Hh + h];
  float4 r; r.x = a.x + d.x; r.y = a.y + d.y; r.z = a.z + d.z; r.w = a.w + d.w;
  *(float4*)&xf[(size_t)row * Hh + h] = r;
}

// ---------------- layernorm (f32 in -> f16 out), one wave per row -------------
__global__ __launch_bounds__(256) void ln_kernel(const float* __restrict__ x,
                                                 const float* __restrict__ g,
                                                 const float* __restrict__ b,
                                                 _Float16* __restrict__ o) {
  int wave = threadIdx.x >> 5, lane = threadIdx.x & 31;
  int row = blockIdx.x * 8 + wave;                  // NROWS % 8 == 0
  const float* xr = x + (size_t)row * Hh;
  float v[8], s = 0.f, sq = 0.f;
#pragma unroll
  for (int t = 0; t < 8; ++t) {
    float f = xr[lane + t * 32];
    v[t] = f; s += f; sq += f * f;
  }
  s = wave_sum(s); sq = wave_sum(sq);
  float mean = s * (1.0f / Hh);
  float var = sq * (1.0f / Hh) - mean * mean;
  float inv = rsqrtf(var + 1e-5f);
  _Float16* orow = o + (size_t)row * Hh;
#pragma unroll
  for (int t = 0; t < 8; ++t) {
    int col = lane + t * 32;
    orow[col] = (_Float16)((v[t] - mean) * inv * g[col] + b[col]);
  }
}

// ---------------- f32 -> f16 cast -------------------------------------------
__global__ __launch_bounds__(256) void cast_kernel(const float* __restrict__ x,
                                                   _Float16* __restrict__ o) {
  int g = blockIdx.x * 256 + threadIdx.x;           // one thread per 4 elems
  if (g >= NROWS * (Hh / 4)) return;
  float4 f = ((const float4*)x)[g];
  union { _Float16 h[4]; uint2 u; } r;
  r.h[0] = (_Float16)f.x; r.h[1] = (_Float16)f.y;
  r.h[2] = (_Float16)f.z; r.h[3] = (_Float16)f.w;
  ((uint2*)o)[g] = r.u;
}

// ---------------- WMMA GEMM: out[M,N] = A[M,K](f16) * W[K,N](f32->f16) -------
// Block tile 128x64, BK=32, 8 waves, each wave 2x2 WMMA 16x16x32 f16 tiles.
// Double-buffered LDS; A tiles staged with global_load_async_to_lds_b128.
// GELU: apply tanh-gelu; RESID: accumulate f32 into outF (residual stream),
// else store f16 into outH. bias==nullptr -> no bias.
template <bool GELU, bool RESID>
__global__ __launch_bounds__(256) void gemm_kernel(const _Float16* __restrict__ A,
                                                   const float* __restrict__ W,
                                                   const float* __restrict__ bias,
                                                   _Float16* __restrict__ outH,
                                                   float* __restrict__ outF,
                                                   int K, int Nn) {
  __shared__ __align__(16) _Float16 As[2][128][40];  // +8 halfs pad, rows 16B aligned
  __shared__ __align__(16) _Float16 Bs[2][64][40];   // stored transposed: Bs[b][n][k]
  const int tid = threadIdx.x;
  const int wave = tid >> 5, lane = tid & 31;
  const int wm = wave & 3, wn = wave >> 2;          // 4x2 wave grid
  const int blockM = blockIdx.x * 128, blockN = blockIdx.y * 64;

  v8f acc[2][2];
#pragma unroll
  for (int mi = 0; mi < 2; ++mi)
#pragma unroll
    for (int ni = 0; ni < 2; ++ni)
      acc[mi][ni] = (v8f){0.f, 0.f, 0.f, 0.f, 0.f, 0.f, 0.f, 0.f};

  const int arow = tid >> 2;          // 0..63
  const int aseg = (tid & 3) * 8;     // 0,8,16,24  (halfs)
  const int bk   = tid >> 3;          // 0..31
  const int bn   = (tid & 7) * 8;     // 0..56

  auto stage = [&](int buf, int k0) {
    // A tile (128x32 f16): async DMA straight into LDS
#pragma unroll
    for (int h = 0; h < 2; ++h) {
      int r = arow + h * 64;
      async_copy_b128(&As[buf][r][aseg],
                      A + (size_t)(blockM + r) * K + k0 + aseg);
    }
    // B tile (32x64 f32 -> f16), transposed into Bs[buf][n][k]
    const float* wrow = W + (size_t)(k0 + bk) * Nn + blockN + bn;
#pragma unroll
    for (int j = 0; j < 8; ++j) Bs[buf][bn + j][bk] = (_Float16)wrow[j];
  };

  stage(0, 0);
  wait_async0();
  __syncthreads();

  const int niter = K / 32;
  for (int it = 0; it < niter; ++it) {
    const int cur = it & 1;
    if (it + 1 < niter) stage(cur ^ 1, (it + 1) * 32);

    AFrag af[2], bf[2];
#pragma unroll
    for (int mi = 0; mi < 2; ++mi) {
      int r = wm * 32 + mi * 16 + (lane & 15);
      int ko = (lane >> 4) * 8;       // lanes 0-15: K0-7/16-23, 16-31: K8-15/24-31
      af[mi].q[0] = *(const uint4*)&As[cur][r][ko];
      af[mi].q[1] = *(const uint4*)&As[cur][r][ko + 16];
    }
#pragma unroll
    for (int ni = 0; ni < 2; ++ni) {
      int cN = wn * 32 + ni * 16 + (lane & 15);
      int ko = (lane >> 4) * 8;       // lanes 0-15: K0-7/16-23, 16-31: K8-15/24-31
      bf[ni].q[0] = *(const uint4*)&Bs[cur][cN][ko];
      bf[ni].q[1] = *(const uint4*)&Bs[cur][cN][ko + 16];
    }
#pragma unroll
    for (int mi = 0; mi < 2; ++mi)
#pragma unroll
      for (int ni = 0; ni < 2; ++ni)
        acc[mi][ni] = __builtin_amdgcn_wmma_f32_16x16x32_f16(
            false, af[mi].h, false, bf[ni].h, (short)0, acc[mi][ni], false, false);

    wait_async0();      // next tile's async DMA landed in LDS
    __syncthreads();    // + this tile's reads done before next overwrite
  }

  // epilogue (C/D layout: VGPR v -> row v or v+8; col = lane&15)
#pragma unroll
  for (int mi = 0; mi < 2; ++mi)
#pragma unroll
    for (int ni = 0; ni < 2; ++ni)
#pragma unroll
      for (int v = 0; v < 8; ++v) {
        int row = blockM + wm * 32 + mi * 16 + v + ((lane >> 4) << 3);
        int col = blockN + wn * 32 + ni * 16 + (lane & 15);
        float val = acc[mi][ni][v];
        if (bias) val += bias[col];
        if (GELU) val = gelu_tanh(val);
        if (RESID) outF[(size_t)row * Nn + col] += val;
        else       outH[(size_t)row * Nn + col] = (_Float16)val;
      }
}

// ---------------- attention core: softmax(Q K^T / sqrt(d)) V per (outer, head)
// MODE 0: attend over pair axis (L=190, row stride C); MODE 1: over column axis
// (L=128, row stride 1). q/k/v/o are [NROWS, H] f16; d = 32 = one lane per dim.
template <int MODE>
__global__ __launch_bounds__(256) void attn_kernel(const _Float16* __restrict__ Q,
                                                   const _Float16* __restrict__ K,
                                                   const _Float16* __restrict__ V,
                                                   _Float16* __restrict__ O) {
  constexpr int L = (MODE == 0) ? Pp : Cc;
  constexpr int MI = (L + 31) / 32;
  __shared__ __align__(16) _Float16 Ks[192][32];
  __shared__ __align__(16) _Float16 Vs[192][32];
  __shared__ float    Ps[8][192];
  __shared__ float    Qs[8][32];

  const int tid = threadIdx.x;
  const int head = blockIdx.x & 7;
  const int outer = blockIdx.x >> 3;
  int base, stride;
  if (MODE == 0) { int b = outer / Cc, c = outer % Cc; base = b * Pp * Cc + c; stride = Cc; }
  else           { int b = outer / Pp, p = outer % Pp; base = (b * Pp + p) * Cc; stride = 1; }

  // stage K,V for this head into LDS (L x 32 f16 each)
  for (int idx = tid; idx < L * 4; idx += 256) {
    int s = idx >> 2, part = (idx & 3) * 8;
    size_t goff = (size_t)(base + s * stride) * Hh + head * Dd + part;
    *(uint4*)&Ks[s][part] = *(const uint4*)(K + goff);
    *(uint4*)&Vs[s][part] = *(const uint4*)(V + goff);
  }
  __syncthreads();

  const int wave = tid >> 5, lane = tid & 31;
  for (int l = wave; l < L; l += 8) {
    size_t qoff = (size_t)(base + l * stride) * Hh + head * Dd;
    Qs[wave][lane] = (float)Q[qoff + lane];
    asm volatile("s_wait_dscnt 0" ::: "memory");

    float s[MI];
    float mx = -1e30f;
#pragma unroll
    for (int mi = 0; mi < MI; ++mi) {
      int m = lane + mi * 32;
      if (m < L) {
        float acc = 0.f;
#pragma unroll
        for (int d = 0; d < Dd; ++d) acc += Qs[wave][d] * (float)Ks[m][d];
        s[mi] = acc * 0.17677669529663687f;   // 1/sqrt(32)
      } else s[mi] = -1e30f;
      mx = fmaxf(mx, s[mi]);
    }
    mx = wave_max(mx);
    float sum = 0.f;
#pragma unroll
    for (int mi = 0; mi < MI; ++mi) { float p = __expf(s[mi] - mx); s[mi] = p; sum += p; }
    sum = wave_sum(sum);
    float inv = 1.0f / sum;
#pragma unroll
    for (int mi = 0; mi < MI; ++mi) {
      int m = lane + mi * 32;
      if (m < L) Ps[wave][m] = s[mi] * inv;
    }
    asm volatile("s_wait_dscnt 0" ::: "memory");

    float acc = 0.f;
    for (int m = 0; m < L; ++m) acc += Ps[wave][m] * (float)Vs[m][lane];
    O[qoff + lane] = (_Float16)acc;
    asm volatile("s_wait_dscnt 0" ::: "memory");
  }
}

// ---------------- head: y1[row] = dot(mid[row,:1024], pW2) + pb2 -------------
__global__ __launch_bounds__(256) void head2_kernel(const _Float16* __restrict__ mid,
                                                    const float* __restrict__ w,
                                                    const float* __restrict__ pb2,
                                                    float* __restrict__ y1) {
  int row = blockIdx.x * 8 + (threadIdx.x >> 5);
  int lane = threadIdx.x & 31;
  const _Float16* mr = mid + (size_t)row * H4;
  float acc = 0.f;
#pragma unroll
  for (int t = 0; t < 32; ++t) { int k = lane + t * 32; acc += (float)mr[k] * w[k]; }
  acc = wave_sum(acc);
  if (lane == 0) y1[row] = acc + pb2[0];
}

// ---------------- final: out[bp] = dot(y1[bp,:C], uW) + ub ------------------
__global__ __launch_bounds__(256) void final_kernel(const float* __restrict__ y1,
                                                    const float* __restrict__ uW,
                                                    const float* __restrict__ ub,
                                                    float* __restrict__ out) {
  int bp = blockIdx.x * 8 + (threadIdx.x >> 5);
  if (bp >= Bc * Pp) return;
  int lane = threadIdx.x & 31;
  float acc = 0.f;
#pragma unroll
  for (int t = 0; t < 4; ++t) { int c = lane + t * 32; acc += y1[bp * Cc + c] * uW[c]; }
  acc = wave_sum(acc);
  if (lane == 0) out[bp] = acc + ub[0];
}

// ---------------- host orchestration ----------------------------------------
extern "C" void kernel_launch(void* const* d_in, const int* in_sizes, int n_in,
                              void* d_out, int out_size, void* d_ws, size_t ws_size,
                              hipStream_t stream) {
  const int*   tokens = (const int*)  d_in[0];
  const float* emb    = (const float*)d_in[1];
  const float* ln1_g  = (const float*)d_in[2];
  const float* ln1_b  = (const float*)d_in[3];
  const float* rWq    = (const float*)d_in[4];
  const float* rWk    = (const float*)d_in[5];
  const float* rWv    = (const float*)d_in[6];
  const float* rWo    = (const float*)d_in[7];
  const float* ln2_g  = (const float*)d_in[8];
  const float* ln2_b  = (const float*)d_in[9];
  const float* cWq    = (const float*)d_in[10];
  const float* cWk    = (const float*)d_in[11];
  const float* cWv    = (const float*)d_in[12];
  const float* cWo    = (const float*)d_in[13];
  const float* ln3_g  = (const float*)d_in[14];
  const float* ln3_b  = (const float*)d_in[15];
  const float* fW1    = (const float*)d_in[16];
  const float* fb1    = (const float*)d_in[17];
  const float* fW2    = (const float*)d_in[18];
  const float* fb2    = (const float*)d_in[19];
  const float* pW1    = (const float*)d_in[20];
  const float* pb1    = (const float*)d_in[21];
  const float* pW2    = (const float*)d_in[22];
  const float* pb2    = (const float*)d_in[23];
  const float* uW     = (const float*)d_in[24];
  const float* ub     = (const float*)d_in[25];
  float* out = (float*)d_out;

  // workspace layout (~249 MB)
  char* w = (char*)d_ws;
  float*     xf  = (float*)w;                                   // [NROWS,H] f32 residual
  _Float16*  hb  = (_Float16*)(w + (size_t)NROWS * Hh * 4);     // [NROWS,H] f16 (LN out / attn out)
  _Float16*  qb  = hb + (size_t)NROWS * Hh;
  _Float16*  kb  = qb + (size_t)NROWS * Hh;
  _Float16*  vb  = kb + (size_t)NROWS * Hh;
  _Float16*  mid = vb + (size_t)NROWS * Hh;                     // [NROWS,4H] f16
  float*     y1  = (float*)(mid + (size_t)NROWS * H4);          // [NROWS]

  const dim3 g256(NROWS / 128, Hh / 64);    // (380, 4)
  const dim3 g1024(NROWS / 128, H4 / 64);   // (380, 16)
  const int ELT4 = NROWS * (Hh / 4) / 256;  // 12160 blocks
  const int LNB  = NROWS / 8;               // 6080 blocks

  embed_kernel<<<ELT4, 256, 0, stream>>>(tokens, emb, xf);

  for (int i = 0; i < 4; ++i) {
    const size_t HH = (size_t)Hh * Hh, HF = (size_t)Hh * H4;
    // --- row attention (over pair axis) ---
    ln_kernel<<<LNB, 256, 0, stream>>>(xf, ln1_g + i * Hh, ln1_b + i * Hh, hb);
    gemm_kernel<false, false><<<g256, 256, 0, stream>>>(hb, rWq + i * HH, nullptr, qb, nullptr, Hh, Hh);
    gemm_kernel<false, false><<<g256, 256, 0, stream>>>(hb, rWk + i * HH, nullptr, kb, nullptr, Hh, Hh);
    gemm_kernel<false, false><<<g256, 256, 0, stream>>>(hb, rWv + i * HH, nullptr, vb, nullptr, Hh, Hh);
    attn_kernel<0><<<Bc * Cc * NHh, 256, 0, stream>>>(qb, kb, vb, hb);
    gemm_kernel<false, true><<<g256, 256, 0, stream>>>(hb, rWo + i * HH, nullptr, nullptr, xf, Hh, Hh);
    // --- column attention ---
    ln_kernel<<<LNB, 256, 0, stream>>>(xf, ln2_g + i * Hh, ln2_b + i * Hh, hb);
    gemm_kernel<false, false><<<g256, 256, 0, stream>>>(hb, cWq + i * HH, nullptr, qb, nullptr, Hh, Hh);
    gemm_kernel<false, false><<<g256, 256, 0, stream>>>(hb, cWk + i * HH, nullptr, kb, nullptr, Hh, Hh);
    gemm_kernel<false, false><<<g256, 256, 0, stream>>>(hb, cWv + i * HH, nullptr, vb, nullptr, Hh, Hh);
    attn_kernel<1><<<Bc * Pp * NHh, 256, 0, stream>>>(qb, kb, vb, hb);
    gemm_kernel<false, true><<<g256, 256, 0, stream>>>(hb, cWo + i * HH, nullptr, nullptr, xf, Hh, Hh);
    // --- FFN ---
    ln_kernel<<<LNB, 256, 0, stream>>>(xf, ln3_g + i * Hh, ln3_b + i * Hh, hb);
    gemm_kernel<true, false><<<g1024, 256, 0, stream>>>(hb, fW1 + i * HF, fb1 + i * H4, mid, nullptr, Hh, H4);
    gemm_kernel<false, true><<<g256, 256, 0, stream>>>(mid, fW2 + i * HF, fb2 + i * Hh, nullptr, xf, H4, Hh);
  }

  // --- head ---
  cast_kernel<<<ELT4, 256, 0, stream>>>(xf, hb);
  gemm_kernel<true, false><<<g1024, 256, 0, stream>>>(hb, pW1, pb1, mid, nullptr, Hh, H4);
  head2_kernel<<<LNB, 256, 0, stream>>>(mid, pW2, pb2, y1);
  final_kernel<<<(Bc * Pp + 7) / 8, 256, 0, stream>>>(y1, uW, ub, out);
  (void)in_sizes; (void)n_in; (void)out_size; (void)ws_size;
}
